// GlobalAttentionLayer_51805895524524
// MI455X (gfx1250) — matrix operand order repaired and checked by reading, not compiled, not run on hardware
//
#include <hip/hip_runtime.h>
#include <hip/hip_bf16.h>

typedef __attribute__((ext_vector_type(8)))  float  v8f;
typedef __attribute__((ext_vector_type(16))) __bf16 v16bf;
typedef __attribute__((ext_vector_type(8)))  __bf16 v8bf;

#define kB   4
#define kL   2048
#define kD   1024
#define kH   4
#define kHD  256
#define kM   (kB * kL)        // 8192 rows
#define kNQ  (3 * kD)         // 3072

__device__ __forceinline__ v8f wmma_bf16(v16bf a, v16bf b, v8f c) {
  return __builtin_amdgcn_wmma_f32_16x16x32_bf16(false, a, false, b, (short)0, c,
                                                 false, false);
}

union V16U { v16bf v; v8bf h[2]; };

// CDNA5 async global->LDS copy (ASYNCcnt-tracked). LDS byte address is the low
// 32 bits of the generic pointer.
__device__ __forceinline__ void async_b128(__bf16* ldsDst, const __bf16* g) {
  unsigned l = (unsigned)(uintptr_t)ldsDst;
  asm volatile("global_load_async_to_lds_b128 %0, %1, off"
               :: "v"(l), "v"(g) : "memory");
}

// ---------------------------------------------------------------- RMSNorm -> bf16
__global__ void k_rmsnorm(const float* __restrict__ x,
                          const float* __restrict__ nw,
                          __bf16* __restrict__ xo) {
  __shared__ float red[256];
  const int row = blockIdx.x;
  const int tid = threadIdx.x;
  const float4 v = ((const float4*)(x + (size_t)row * kD))[tid];
  float ss = v.x * v.x + v.y * v.y + v.z * v.z + v.w * v.w;
  red[tid] = ss;
  __syncthreads();
  for (int s = 128; s > 0; s >>= 1) {
    if (tid < s) red[tid] += red[tid + s];
    __syncthreads();
  }
  const float inv = rsqrtf(red[0] * (1.0f / kD) + 1e-6f);
  const float4 w = ((const float4*)nw)[tid];
  __bf16* o = xo + (size_t)row * kD + tid * 4;
  o[0] = (__bf16)(v.x * inv * w.x);
  o[1] = (__bf16)(v.y * inv * w.y);
  o[2] = (__bf16)(v.z * inv * w.z);
  o[3] = (__bf16)(v.w * inv * w.w);
}

// ---------------------------------------------------------------- f32 -> bf16 convert
__global__ void k_cvt(const float* __restrict__ w, __bf16* __restrict__ o, int n) {
  int i = blockIdx.x * blockDim.x + threadIdx.x;
  if (i < n) o[i] = (__bf16)w[i];
}

// ---------------------------------------------------------------- GEMM: C[M,N] = A[M,K] @ W[N,K]^T
// block = 256 thr (8 waves) arranged 2(M) x 4(N): block tile 32x256.
// B panel (256 rows x 32 K, bf16) double-buffered in LDS via async DMA.
// grid = (N/256, M/32)
__global__ void k_gemm(const __bf16* __restrict__ A, const __bf16* __restrict__ W,
                       float* __restrict__ C, int N, int K,
                       const float* __restrict__ residual,
                       const float* __restrict__ resScale) {
  __shared__ __bf16 bpanel[2][256 * 32];   // 2 x 16 KB

  const int tid  = threadIdx.x;
  const int lane = tid & 31;
  const int wave = tid >> 5;
  const int half = lane >> 4;
  const int col  = lane & 15;

  const int mBase = blockIdx.y * 32 + (wave >> 2) * 16;  // wave's 16-row A tile
  const int nBlk  = blockIdx.x * 256;                    // block's B panel base
  const int nLoc  = (wave & 3) * 64;                     // wave's N offset in panel

  // stage one 256x32 B panel: each thread owns one row (64 B = 4 x b128)
  auto stage = [&](__bf16* buf, int kc) {
    const __bf16* g = W + (size_t)(nBlk + tid) * K + kc;
    __bf16* l = buf + tid * 32;
#pragma unroll
    for (int p = 0; p < 4; ++p) async_b128(l + p * 8, g + p * 8);
  };

  v8f acc[4] = {};
  const int aRow = mBase + col;
  const int nIter = K / 32;

  stage(bpanel[0], 0);

  for (int it = 0; it < nIter; ++it) {
    if (it + 1 < nIter) {
      stage(bpanel[(it + 1) & 1], (it + 1) * 32);
      asm volatile("s_wait_asynccnt 4" ::: "memory");  // batch `it` landed
    } else {
      asm volatile("s_wait_asynccnt 0" ::: "memory");
    }
    __syncthreads();  // all waves' DMA for buf[it&1] visible

    const int kc = it * 32;
    V16U a;
    const __bf16* ap = A + (size_t)aRow * K + kc + half * 8;
    a.h[0] = *(const v8bf*)ap;
    a.h[1] = *(const v8bf*)(ap + 16);

    const __bf16* bp = bpanel[it & 1];
#pragma unroll
    for (int t = 0; t < 4; ++t) {
      V16U b;
      const __bf16* lp = bp + (size_t)(nLoc + t * 16 + col) * 32 + half * 8;
      b.h[0] = *(const v8bf*)lp;
      b.h[1] = *(const v8bf*)(lp + 16);
      acc[t] = wmma_bf16(a.v, b.v, acc[t]);
    }
    __syncthreads();  // nobody still reads buf[it&1] when stage(it+2) rewrites it
  }

  const float rs = residual ? *resScale : 0.0f;
#pragma unroll
  for (int t = 0; t < 4; ++t) {
#pragma unroll
    for (int i = 0; i < 8; ++i) {
      const int r = mBase + i + half * 8;
      const int c = nBlk + nLoc + t * 16 + col;
      const size_t idx = (size_t)r * N + c;
      float v = acc[t][i];
      C[idx] = residual ? (residual[idx] + rs * v) : v;
    }
  }
}

// ---------------------------------------------------------------- RoPE + layout reorg
// qkv f32 [M, 3D] -> q,k bf16 [B,H,L,hd]; v bf16 transposed [B,H,hd,L]
__global__ void k_rope(const float* __restrict__ qkv, __bf16* __restrict__ qo,
                       __bf16* __restrict__ ko, __bf16* __restrict__ vt) {
  const int idx = blockIdx.x;         // bh * L + l
  const int bh = idx >> 11;
  const int l  = idx & (kL - 1);
  const int b  = bh >> 2;
  const int h  = bh & 3;
  const int d  = threadIdx.x;         // 0..255
  const size_t rowOff = ((size_t)(b * kL + l)) * kNQ;

  const float qv = qkv[rowOff + h * kHD + d];
  const float kv = qkv[rowOff + kD + h * kHD + d];
  const float vv = qkv[rowOff + 2 * kD + h * kHD + d];

  const int j = d & 127;
  // inv_freq = 10000^(-j/128) = exp2(-j * log2(10000)/128)
  const float ang = (float)l * exp2f(-(float)j * 0.10381025296523008f);
  const float sn = __sinf(ang);
  const float cs = __cosf(ang);
  const int   dp  = (d < 128) ? d + 128 : d - 128;
  const float sgn = (d < 128) ? -1.0f : 1.0f;
  const float q2 = qkv[rowOff + h * kHD + dp];
  const float k2 = qkv[rowOff + kD + h * kHD + dp];

  const size_t o1 = ((size_t)bh * kL + l) * kHD + d;
  qo[o1] = (__bf16)(qv * cs + sgn * q2 * sn);
  ko[o1] = (__bf16)(kv * cs + sgn * k2 * sn);
  vt[((size_t)bh * kHD + d) * kL + l] = (__bf16)vv;
}

// ---------------------------------------------------------------- Flash attention
// block = 128 thr (4 waves); wave -> one 16-row Q tile. grid = B*H*(L/16)/4 = 512
__global__ void k_attn(const __bf16* __restrict__ Q, const __bf16* __restrict__ Kp,
                       const __bf16* __restrict__ Vt, __bf16* __restrict__ O) {
  __shared__ float  s_lds[4][16 * 32];
  __shared__ __bf16 p_lds[4][16 * 32];
  __shared__ float  a_lds[4][16];
  __shared__ float  r_lds[4][16];

  const int lane = threadIdx.x & 31;
  const int wave = threadIdx.x >> 5;
  const int tile = blockIdx.x * 4 + wave;
  const int bh = tile >> 7;           // 128 q-tiles per (b,h)
  const int qt = tile & 127;
  const int half = lane >> 4;
  const int col  = lane & 15;

  const __bf16* Qbh = Q  + (size_t)bh * kL * kHD;
  const __bf16* Kbh = Kp + (size_t)bh * kL * kHD;
  const __bf16* Vbh = Vt + (size_t)bh * kHD * kL;
  const int qrow = qt * 16 + col;

  v8f Ov[16] = {};
  float m = -1e30f, lsum = 0.0f;

  for (int kb = 0; kb < kL; kb += 32) {
    // ---- S = (Q K^T) * hd^-0.5 over this 16x32 key tile
    v8f S0 = {}, S1 = {};
#pragma unroll
    for (int kc = 0; kc < 8; ++kc) {
      V16U a;
      const __bf16* qp = Qbh + (size_t)qrow * kHD + kc * 32 + half * 16;
      a.h[0] = *(const v8bf*)qp;
      a.h[1] = *(const v8bf*)(qp + 8);
      v16bf b0 = *(const v16bf*)(Kbh + (size_t)(kb + col) * kHD + kc * 32 + half * 16);
      v16bf b1 = *(const v16bf*)(Kbh + (size_t)(kb + 16 + col) * kHD + kc * 32 + half * 16);
      S0 = wmma_bf16(a.v, b0, S0);
      S1 = wmma_bf16(a.v, b1, S1);
    }
#pragma unroll
    for (int i = 0; i < 8; ++i) { S0[i] *= 0.0625f; S1[i] *= 0.0625f; }

    // ---- spill C-layout S to LDS for row-wise softmax
#pragma unroll
    for (int i = 0; i < 8; ++i) {
      s_lds[wave][(i + half * 8) * 32 + col]      = S0[i];
      s_lds[wave][(i + half * 8) * 32 + 16 + col] = S1[i];
    }
    asm volatile("s_wait_dscnt 0" ::: "memory");

    // ---- online softmax stats: lane r < 16 owns row r
    if (lane < 16) {
      float vals[32];
      float mnew = m;
#pragma unroll
      for (int c = 0; c < 32; ++c) {
        vals[c] = s_lds[wave][lane * 32 + c];
        mnew = fmaxf(mnew, vals[c]);
      }
      const float alpha = __expf(m - mnew);
      float sum = 0.0f;
#pragma unroll
      for (int c = 0; c < 32; ++c) {
        const float p = __expf(vals[c] - mnew);
        sum += p;
        p_lds[wave][lane * 32 + c] = (__bf16)p;
      }
      lsum = lsum * alpha + sum;
      m = mnew;
      a_lds[wave][lane] = alpha;
    }
    asm volatile("s_wait_dscnt 0" ::: "memory");

    // ---- rescale O by per-row alpha
    float af[8];
#pragma unroll
    for (int i = 0; i < 8; ++i) af[i] = a_lds[wave][i + half * 8];
#pragma unroll
    for (int nt = 0; nt < 16; ++nt)
#pragma unroll
      for (int i = 0; i < 8; ++i) Ov[nt][i] *= af[i];

    // ---- O += P @ V : load P in A-layout from LDS, V^T columns contiguous
    V16U P;
    const __bf16* pp = &p_lds[wave][col * 32 + half * 16];
    P.h[0] = *(const v8bf*)pp;
    P.h[1] = *(const v8bf*)(pp + 8);
#pragma unroll
    for (int nt = 0; nt < 16; ++nt) {
      v16bf bv = *(const v16bf*)(Vbh + (size_t)(nt * 16 + col) * kL + kb + half * 16);
      Ov[nt] = wmma_bf16(P.v, bv, Ov[nt]);
    }
  }

  if (lane < 16) r_lds[wave][lane] = 1.0f / lsum;
  asm volatile("s_wait_dscnt 0" ::: "memory");
  float rf[8];
#pragma unroll
  for (int i = 0; i < 8; ++i) rf[i] = r_lds[wave][i + half * 8];

  const int b = bh >> 2, h = bh & 3;
#pragma unroll
  for (int nt = 0; nt < 16; ++nt)
#pragma unroll
    for (int i = 0; i < 8; ++i) {
      const int r = qt * 16 + i + half * 8;
      O[((size_t)(b * kL + r)) * kD + h * kHD + nt * 16 + col] =
          (__bf16)(Ov[nt][i] * rf[i]);
    }
}

// ---------------------------------------------------------------- launch
extern "C" void kernel_launch(void* const* d_in, const int* in_sizes, int n_in,
                              void* d_out, int out_size, void* d_ws, size_t ws_size,
                              hipStream_t stream) {
  const float* x        = (const float*)d_in[0];
  const float* norm_w   = (const float*)d_in[1];
  const float* qkv_w    = (const float*)d_in[2];
  const float* out_w    = (const float*)d_in[3];
  const float* resScale = (const float*)d_in[4];

  char* ws = (char*)d_ws;
  size_t off = 0;
  auto alloc = [&](size_t bytes) {
    size_t p = off;
    off += (bytes + 255) & ~(size_t)255;
    return p;
  };
  __bf16* xnbf   = (__bf16*)(ws + alloc((size_t)kM * kD * 2));
  __bf16* qkvwbf = (__bf16*)(ws + alloc((size_t)kNQ * kD * 2));
  __bf16* outwbf = (__bf16*)(ws + alloc((size_t)kD * kD * 2));
  __bf16* qbf    = (__bf16*)(ws + alloc((size_t)kM * kD * 2));
  __bf16* kbf    = (__bf16*)(ws + alloc((size_t)kM * kD * 2));
  __bf16* vbf    = (__bf16*)(ws + alloc((size_t)kM * kD * 2));
  __bf16* aobf   = (__bf16*)(ws + alloc((size_t)kM * kD * 2));
  float*  qkvf   = (float*)(ws + alloc((size_t)kM * kNQ * 4));
  (void)ws_size; (void)in_sizes; (void)n_in; (void)out_size;

  // 1) RMSNorm + bf16
  k_rmsnorm<<<kM, 256, 0, stream>>>(x, norm_w, xnbf);
  // 2) weight converts
  k_cvt<<<(kNQ * kD) / 256, 256, 0, stream>>>(qkv_w, qkvwbf, kNQ * kD);
  k_cvt<<<(kD * kD) / 256, 256, 0, stream>>>(out_w, outwbf, kD * kD);
  // 3) fused QKV projection (WMMA + async LDS staging)
  k_gemm<<<dim3(kNQ / 256, kM / 32), 256, 0, stream>>>(xnbf, qkvwbf, qkvf, kNQ, kD,
                                                       nullptr, nullptr);
  // 4) RoPE + reorg (q,k row-major per head; v transposed)
  k_rope<<<kB * kH * kL, kHD, 0, stream>>>(qkvf, qbf, kbf, vbf);
  // 5) flash attention (WMMA QK^T + PV, online softmax)
  k_attn<<<(kB * kH * (kL / 16)) / 4, 128, 0, stream>>>(qbf, kbf, vbf, aobf);
  // 6) out projection + residual epilogue (WMMA + async LDS staging)
  k_gemm<<<dim3(kD / 256, kM / 32), 256, 0, stream>>>(aobf, outwbf, (float*)d_out,
                                                      kD, kD, x, resScale);
}
